// SelfAttention_38448547233995
// MI455X (gfx1250) — matrix operand order repaired and checked by reading, not compiled, not run on hardware
//
#include <hip/hip_runtime.h>
#include <hip/hip_bf16.h>

typedef __attribute__((ext_vector_type(16))) _Float16 v16h;
typedef __attribute__((ext_vector_type(8)))  float    v8f;
typedef __attribute__((ext_vector_type(4)))  int      v4i;

#define WMMA_F16(a, b, c) \
  __builtin_amdgcn_wmma_f32_16x16x32_f16(false, (a), false, (b), (short)0, (c), false, false)

// ---- async global->LDS staging (gfx1250 GLOBAL_LOAD_ASYNC_TO_LDS_B128) ----
#if defined(__has_builtin)
#if __has_builtin(__builtin_amdgcn_global_load_async_to_lds_b128)
#define HAVE_ASYNC_LDS 1
#endif
#endif
#ifndef HAVE_ASYNC_LDS
#define HAVE_ASYNC_LDS 0
#endif

__device__ __forceinline__ void stage_b128(void* lds_dst, const void* gsrc) {
#if HAVE_ASYNC_LDS
  __builtin_amdgcn_global_load_async_to_lds_b128(
      (__attribute__((address_space(1))) v4i*)gsrc,
      (__attribute__((address_space(3))) v4i*)lds_dst, 0, 0);
#else
  *(uint4*)lds_dst = *(const uint4*)gsrc;
#endif
}

__device__ __forceinline__ void stage_wait() {
#if HAVE_ASYNC_LDS
#if defined(__has_builtin) && __has_builtin(__builtin_amdgcn_s_wait_asynccnt)
  __builtin_amdgcn_s_wait_asynccnt(0);
#else
  asm volatile("s_wait_asynccnt 0x0" ::: "memory");
#endif
#endif
}

union Frag16 { v16h v; _Float16 h[16]; unsigned u[8]; };

// A-matrix 16x32 f16 fragment. rowp = this lane's row M, contiguous K halves.
// element e -> K = e + 16*group   (group = lane>=16)
__device__ __forceinline__ v16h frag_a(const _Float16* rowp, int group) {
  Frag16 f;
  const unsigned* p0 = (const unsigned*)(rowp + 16 * group);
  const unsigned* p1 = (const unsigned*)(rowp + 16 * group + 8);
  f.u[0] = p0[0]; f.u[1] = p0[1]; f.u[2] = p0[2]; f.u[3] = p0[3];
  f.u[4] = p1[0]; f.u[5] = p1[1]; f.u[6] = p1[2]; f.u[7] = p1[3];
  return f.v;
}

// B-matrix 32x16 f16 fragment. rowp = this lane's column N, contiguous K halves.
// element e -> K = e + 16*group
__device__ __forceinline__ v16h frag_b(const _Float16* rowp, int group) {
  Frag16 f;
  const unsigned* p = (const unsigned*)(rowp + 16 * group);
#pragma unroll
  for (int j = 0; j < 8; ++j) f.u[j] = p[j];
  return f.v;
}

// B-matrix fragment from a K-major (strided) source: base = &src[K=0][this lane's N].
__device__ __forceinline__ v16h frag_b_strided(const _Float16* base, int ldk, int group) {
  Frag16 f;
#pragma unroll
  for (int e = 0; e < 16; ++e) f.h[e] = base[(e + 16 * group) * ldk];
  return f.v;
}

// ---------------------------------------------------------------------------
// Kernel 0: f32 -> f16 conversion (x is read by all three GEMMs; converting
// once cuts A-side HBM traffic in half and enables async byte-copy staging).
// ---------------------------------------------------------------------------
__global__ __launch_bounds__(256) void cvt_f16_kernel(
    const float* __restrict__ src, _Float16* __restrict__ dst) {
  size_t i = ((size_t)blockIdx.x * 256 + threadIdx.x) * 4;
  float4 v = *(const float4*)(src + i);
  union { _Float16 h[4]; uint2 u; } p;
  p.h[0] = (_Float16)v.x; p.h[1] = (_Float16)v.y;
  p.h[2] = (_Float16)v.z; p.h[3] = (_Float16)v.w;
  *(uint2*)(dst + i) = p.u;
}

// ---------------------------------------------------------------------------
// Kernel 1: C[M=8192, N=1024] = Xh[8192,1024](f16) * W[1024,1024](f32) + bias.
// Block tile 128x64, K-step 32, 256 threads = 8 waves in a 4x2 wave grid.
// A tile staged via async global->LDS; B tile converted+transposed manually.
// ---------------------------------------------------------------------------
__global__ __launch_bounds__(256) void qkv_gemm_kernel(
    const _Float16* __restrict__ Xh, const float* __restrict__ W,
    const float* __restrict__ bias, _Float16* __restrict__ out) {
  __shared__ _Float16 As[128 * 40];  // rows x K (pad 32->40 halves, 80B rows)
  __shared__ _Float16 Bs[64 * 40];   // cols(N) x K, stored transposed

  const int tid  = threadIdx.x;
  const int w    = tid >> 5, lane = tid & 31;
  const int g    = lane >> 4, r = lane & 15;
  const int wm   = w & 3, wn = w >> 2;           // 4x2 waves -> 32x32 each
  const int m0   = blockIdx.y * 128;
  const int n0   = blockIdx.x * 64;

  v8f acc[2][2] = {};

  for (int k0 = 0; k0 < 1024; k0 += 32) {
    __syncthreads();
    // Stage A tile: 128 rows x 32 halves = 512 b128 chunks (2 per thread)
#pragma unroll
    for (int i = 0; i < 2; ++i) {
      int idx = tid + i * 256;
      int row = idx >> 2, c = idx & 3;
      stage_b128(As + row * 40 + c * 8,
                 Xh + (size_t)(m0 + row) * 1024 + k0 + c * 8);
    }
    // Stage B tile transposed: W[k0..k0+32][n0..n0+64] f32 -> Bs[n][k] f16
#pragma unroll
    for (int i = 0; i < 2; ++i) {
      int idx = tid + i * 256;
      int kr = idx >> 4, c4 = idx & 15;
      float4 wv = *(const float4*)(W + (size_t)(k0 + kr) * 1024 + n0 + c4 * 4);
      Bs[(c4 * 4 + 0) * 40 + kr] = (_Float16)wv.x;
      Bs[(c4 * 4 + 1) * 40 + kr] = (_Float16)wv.y;
      Bs[(c4 * 4 + 2) * 40 + kr] = (_Float16)wv.z;
      Bs[(c4 * 4 + 3) * 40 + kr] = (_Float16)wv.w;
    }
    stage_wait();
    __syncthreads();

    v16h a0 = frag_a(&As[(wm * 32 + r) * 40], g);
    v16h a1 = frag_a(&As[(wm * 32 + 16 + r) * 40], g);
    v16h b0 = frag_b(&Bs[(wn * 32 + r) * 40], g);
    v16h b1 = frag_b(&Bs[(wn * 32 + 16 + r) * 40], g);
    acc[0][0] = WMMA_F16(a0, b0, acc[0][0]);
    acc[0][1] = WMMA_F16(a0, b1, acc[0][1]);
    acc[1][0] = WMMA_F16(a1, b0, acc[1][0]);
    acc[1][1] = WMMA_F16(a1, b1, acc[1][1]);
  }

  // Epilogue: +bias, f16 store. C layout: VGPR i -> row 8*g+i, lane%16 -> col.
#pragma unroll
  for (int mt = 0; mt < 2; ++mt)
#pragma unroll
    for (int nt = 0; nt < 2; ++nt) {
      int nc = n0 + wn * 32 + nt * 16 + r;
      float bv = bias[nc];
#pragma unroll
      for (int i = 0; i < 8; ++i) {
        int mr = m0 + wm * 32 + mt * 16 + 8 * g + i;
        out[(size_t)mr * 1024 + nc] = (_Float16)(acc[mt][nt][i] + bv);
      }
    }
}

// ---------------------------------------------------------------------------
// Kernel 2: flash attention. One workgroup = 32 query rows of one batch.
// 8 waves: S phase -> wave (qi = w&1 row-tile, kj = w>>1 col-tile) of S(32x64);
// PV phase -> each wave owns O columns [w*128, w*128+128) (16 f32 accums).
// K/V/Q blocks staged with async global->LDS (no VGPR round trip).
// ---------------------------------------------------------------------------
#define KV_STRIDE 1032   // 1024 + 8 halves (2064B rows keep 16B alignment)
#define SMEM_BYTES (64 * KV_STRIDE * 2 + 32 * KV_STRIDE * 2 + 32 * 72 * 2 + 32 * 65 * 4 + 3 * 32 * 4)

__global__ __launch_bounds__(256) void attn_kernel(
    const _Float16* __restrict__ Qh, const _Float16* __restrict__ Kh,
    const _Float16* __restrict__ Vh, float* __restrict__ out) {
  extern __shared__ char smem[];
  _Float16* KVs  = (_Float16*)smem;                 // [64][KV_STRIDE] K then V
  _Float16* Qs   = KVs + 64 * KV_STRIDE;            // [32][KV_STRIDE]
  _Float16* Pb   = Qs + 32 * KV_STRIDE;             // [32][72] softmax probs f16
  float*    Sb   = (float*)(Pb + 32 * 72);          // [32][65] raw scores f32
  float*    mrow = Sb + 32 * 65;                    // running max
  float*    lrow = mrow + 32;                       // running denom
  float*    crow = lrow + 32;                       // per-step rescale factor

  const int tid = threadIdx.x;
  const int w = tid >> 5, lane = tid & 31, g = lane >> 4, r = lane & 15;
  const int b = blockIdx.y, q0 = blockIdx.x * 32;
  const size_t ND = (size_t)2048 * 1024;
  const _Float16* Qg = Qh + b * ND + (size_t)q0 * 1024;
  const _Float16* Kg = Kh + b * ND;
  const _Float16* Vg = Vh + b * ND;

  // Stage Q block (32x1024 halves = 4096 b128 chunks); waited with first K wait
#pragma unroll
  for (int i = 0; i < 16; ++i) {
    int idx = tid + i * 256;
    int row = idx >> 7, c = idx & 127;
    stage_b128(Qs + row * KV_STRIDE + c * 8, Qg + (size_t)row * 1024 + c * 8);
  }
  if (tid < 32) { mrow[tid] = -3.0e38f; lrow[tid] = 0.0f; }

  v8f acc[2][8] = {};              // O slice: rows 32, cols [w*128, +128)
  const int qi = w & 1, kj = w >> 1;

  for (int jb = 0; jb < 32; ++jb) {
    __syncthreads();               // prior-iteration PV reads of KVs done
    // Stage K block (64x1024 halves = 8192 b128 chunks)
    const _Float16* ksrc = Kg + (size_t)jb * 64 * 1024;
#pragma unroll
    for (int i = 0; i < 32; ++i) {
      int idx = tid + i * 256;
      int row = idx >> 7, c = idx & 127;
      stage_b128(KVs + row * KV_STRIDE + c * 8, ksrc + (size_t)row * 1024 + c * 8);
    }
    stage_wait();
    __syncthreads();

    // S tile = Q(16x1024) . K^T(1024x16), reduce over d in 32-steps
    v8f s = {};
    const _Float16* qrow = &Qs[(qi * 16 + r) * KV_STRIDE];
    const _Float16* krow = &KVs[(kj * 16 + r) * KV_STRIDE];
#pragma unroll 4
    for (int d = 0; d < 1024; d += 32) {
      v16h a  = frag_a(qrow + d, g);
      v16h bb = frag_b(krow + d, g);   // row of K == column of K^T
      s = WMMA_F16(a, bb, s);
    }
#pragma unroll
    for (int i = 0; i < 8; ++i)
      Sb[(qi * 16 + 8 * g + i) * 65 + kj * 16 + r] = s[i] * 0.03125f; // 1/sqrt(1024)
    __syncthreads();

    // Online softmax: 8 threads per row, 8 cols each
    {
      int row = tid >> 3, sub = tid & 7;
      float* sp = &Sb[row * 65 + sub * 8];
      float mx = sp[0];
#pragma unroll
      for (int c = 1; c < 8; ++c) mx = fmaxf(mx, sp[c]);
      mx = fmaxf(mx, __shfl_xor(mx, 1, 8));
      mx = fmaxf(mx, __shfl_xor(mx, 2, 8));
      mx = fmaxf(mx, __shfl_xor(mx, 4, 8));
      float mold = mrow[row];
      float mnew = fmaxf(mold, mx);
      float cfac = __expf(mold - mnew);
      float ps = 0.0f;
#pragma unroll
      for (int c = 0; c < 8; ++c) {
        float p = __expf(sp[c] - mnew);
        Pb[row * 72 + sub * 8 + c] = (_Float16)p;
        ps += p;
      }
      ps += __shfl_xor(ps, 1, 8);
      ps += __shfl_xor(ps, 2, 8);
      ps += __shfl_xor(ps, 4, 8);
      if (sub == 0) { mrow[row] = mnew; crow[row] = cfac; lrow[row] = lrow[row] * cfac + ps; }
    }
    __syncthreads();

    // Rescale register-resident O by exp(m_old - m_new)
    float cf[2][8];
#pragma unroll
    for (int mt = 0; mt < 2; ++mt)
#pragma unroll
      for (int i = 0; i < 8; ++i) cf[mt][i] = crow[mt * 16 + 8 * g + i];
#pragma unroll
    for (int mt = 0; mt < 2; ++mt)
#pragma unroll
      for (int nt = 0; nt < 8; ++nt)
#pragma unroll
        for (int i = 0; i < 8; ++i) acc[mt][nt][i] *= cf[mt][i];

    // Stage V block into the same LDS buffer (K no longer needed)
    const _Float16* vsrc = Vg + (size_t)jb * 64 * 1024;
#pragma unroll
    for (int i = 0; i < 32; ++i) {
      int idx = tid + i * 256;
      int row = idx >> 7, c = idx & 127;
      stage_b128(KVs + row * KV_STRIDE + c * 8, vsrc + (size_t)row * 1024 + c * 8);
    }
    stage_wait();
    __syncthreads();

    // O += P(32x64) . V(64x128-slice): K-dim 64 in two 32-chunks
    const int ncol = w * 128;
#pragma unroll
    for (int kc = 0; kc < 64; kc += 32) {
      v16h a0 = frag_a(&Pb[r * 72 + kc], g);
      v16h a1 = frag_a(&Pb[(16 + r) * 72 + kc], g);
#pragma unroll
      for (int nt = 0; nt < 8; ++nt) {
        v16h bb = frag_b_strided(&KVs[(size_t)kc * KV_STRIDE + ncol + nt * 16 + r],
                                 KV_STRIDE, g);
        acc[0][nt] = WMMA_F16(a0, bb, acc[0][nt]);
        acc[1][nt] = WMMA_F16(a1, bb, acc[1][nt]);
      }
    }
  }
  __syncthreads();

  // Epilogue: O / l, f32 store
  float inv[2][8];
#pragma unroll
  for (int mt = 0; mt < 2; ++mt)
#pragma unroll
    for (int i = 0; i < 8; ++i) inv[mt][i] = 1.0f / lrow[mt * 16 + 8 * g + i];
#pragma unroll
  for (int mt = 0; mt < 2; ++mt)
#pragma unroll
    for (int nt = 0; nt < 8; ++nt)
#pragma unroll
      for (int i = 0; i < 8; ++i) {
        size_t row = (size_t)(b * 2048 + q0 + mt * 16 + 8 * g + i);
        out[row * 1024 + w * 128 + nt * 16 + r] = acc[mt][nt][i] * inv[mt][i];
      }
}

// ---------------------------------------------------------------------------
extern "C" void kernel_launch(void* const* d_in, const int* in_sizes, int n_in,
                              void* d_out, int out_size, void* d_ws, size_t ws_size,
                              hipStream_t stream) {
  (void)in_sizes; (void)n_in; (void)out_size; (void)ws_size;
  const float* x  = (const float*)d_in[0];
  const float* Wq = (const float*)d_in[1];
  const float* Wk = (const float*)d_in[2];
  const float* Wv = (const float*)d_in[3];
  const float* bq = (const float*)d_in[4];
  const float* bk = (const float*)d_in[5];
  const float* bv = (const float*)d_in[6];
  float* out = (float*)d_out;

  const size_t BND = (size_t)4 * 2048 * 1024;   // 8.39M elems per tensor
  _Float16* Qh = (_Float16*)d_ws;
  _Float16* Kh = Qh + BND;
  _Float16* Vh = Kh + BND;
  _Float16* Xh = Vh + BND;                      // 64 MB f16 workspace total

  // x f32 -> f16 once (read 3x by the projection GEMMs)
  cvt_f16_kernel<<<dim3(BND / (256 * 4)), 256, 0, stream>>>(x, Xh);

  dim3 g1(1024 / 64, 8192 / 128);               // (16, 64)
  qkv_gemm_kernel<<<g1, 256, 0, stream>>>(Xh, Wq, bq, Qh);
  qkv_gemm_kernel<<<g1, 256, 0, stream>>>(Xh, Wk, bk, Kh);
  qkv_gemm_kernel<<<g1, 256, 0, stream>>>(Xh, Wv, bv, Vh);

  dim3 g2(2048 / 32, 4);                        // (64, 4) workgroups
  attn_kernel<<<g2, 256, SMEM_BYTES, stream>>>(Qh, Kh, Vh, out);
}
